// regionAgg_layer_42116449304773
// MI455X (gfx1250) — matrix operand order repaired and checked by reading, not compiled, not run on hardware
//
#include <hip/hip_runtime.h>

// Problem constants (from the reference)
#define RREG   1024
#define PPIX   131072
#define DDIM   64

// Tiling
#define BM     128                 // regions per workgroup: 8 x 16 M-tiles
#define BK     32                  // pixels (K) per main-loop iteration
#define KSPLIT 64                  // split-K -> 8*64 = 512 workgroups
#define KRANGE (PPIX / KSPLIT)     // 2048 pixels per workgroup
#define NITER  (KRANGE / BK)       // 64 iterations

// Operand-ready bf16 LDS layout: each of the 32 lanes owns a contiguous 32B
// slot (16 halves in WMMA VGPR order), padded to a 48B lane stride so b128
// ops sit at the inherent 2-way bank-conflict floor.
#define LSTR   24                  // halves per lane slot (48B)
#define TSZ    (32 * LSTR)         // halves per operand tile

typedef __bf16 bf16_t;
typedef __attribute__((ext_vector_type(16))) __bf16 v16bf;
typedef __attribute__((ext_vector_type(8)))  __bf16 v8bf;
typedef __attribute__((ext_vector_type(4)))  __bf16 v4bf;
typedef __attribute__((ext_vector_type(8)))  float  v8f;
typedef __attribute__((ext_vector_type(4)))  float  v4f;
typedef __attribute__((ext_vector_type(2)))  float  v2f;

__device__ __forceinline__ v16bf ldsOperand(const bf16_t* p) {
  v8bf lo = *(const v8bf*)(p);
  v8bf hi = *(const v8bf*)(p + 8);
  return __builtin_shufflevector(lo, hi, 0,1,2,3,4,5,6,7,8,9,10,11,12,13,14,15);
}

// d_out is poisoned by the harness -> zero it before atomic accumulation.
__global__ __launch_bounds__(128) void regionAgg_zero(float* __restrict__ out) {
  v4f z = {0.f, 0.f, 0.f, 0.f};
  ((v4f*)out)[blockIdx.x * 128 + threadIdx.x] = z;
}

__global__ __launch_bounds__(128) void regionAgg_wmma(
    const float* __restrict__ mask,   // [RREG, PPIX] binary one-hot
    const float* __restrict__ x,      // [PPIX, DDIM]
    float* __restrict__ out) {        // [RREG, DDIM]
  __shared__ bf16_t sA[8 * TSZ];      // 8 A tiles (16x32 bf16, operand-ready)
  __shared__ bf16_t sB[4 * TSZ];      // 4 B tiles (32x16 bf16, operand-ready)

  const int t    = threadIdx.x;       // 0..127 (4 waves)
  const int lane = t & 31;
  const int wave = t >> 5;
  const int mb   = blockIdx.x / KSPLIT;
  const int ks   = blockIdx.x % KSPLIT;
  const int m0   = mb * BM;
  const size_t kbeg = (size_t)ks * KRANGE;

  const int l15 = lane & 15;
  const int hi  = lane >> 4;

  // Square wave blocking: wave -> (M half mh: 4 tiles) x (N half nh: 2 tiles)
  const int nh = wave & 1;            // columns 32*nh .. +31
  const int mh = wave >> 1;           // rows    64*mh .. +63 (within block)

  // ---- A staging map: thread -> (rows ar0+16i, k = ac..ac+3) ----
  // A VGPR layout: lanes 0-15: K {0..7}u{16..23}; lanes 16-31: K {8..15}u{24..31}
  const int ac  = (t & 7) * 4;                          // k offset, bit3 const
  const int ar0 = t >> 3;                               // row-in-tile 0..15
  const int aL  = ar0 + (((ac >> 3) & 1) ? 16 : 0);     // dest lane slot
  const int ae0 = (ac & 7) + ((ac >= 16) ? 8 : 0);      // dest elem 0,4,8,12

  // ---- B staging map: thread -> (columns 2np, 2np+1; k-octet kg8*8..+7) ----
  // B VGPR layout: lanes 0-15: K {0..7}u{16..23}; lanes 16-31: K {8..15}u{24..31}
  const int np  = t & 31;                               // column pair index
  const int kg8 = t >> 5;                               // 0..3
  const int be0 = 8 * (kg8 >> 1);                       // elem offset 0 or 8
  const int bLo = ((2 * np) & 15) + ((kg8 & 1) ? 16 : 0);
  bf16_t* sBp0 = sB + ((2 * np) >> 4) * TSZ + bLo * LSTR + be0;
  bf16_t* sBp1 = sBp0 + LSTR;                           // adjacent column slot

  v8f acc[8] = {};                    // [a-tile 0..3][n-tile 0..1]
  v4f  rawA[8];                       // in-flight f32 mask rows
  v2f  rawB[8];                       // in-flight f32 x column pairs
  v4bf hA[8];                         // converted, store-ready
  v8bf hB0, hB1;

  auto loadTiles = [&](int it) {
    const size_t kofs = kbeg + (size_t)it * BK;
    const float* mrow = mask + (size_t)m0 * PPIX + kofs + ac;
#pragma unroll
    for (int i = 0; i < 8; ++i)
      rawA[i] = __builtin_nontemporal_load(
          (const v4f*)(mrow + (size_t)(ar0 + 16 * i) * PPIX));
    const float* xrow = x + (kofs + 8 * kg8) * DDIM + 2 * np;
#pragma unroll
    for (int j = 0; j < 8; ++j) rawB[j] = *(const v2f*)(xrow + j * DDIM);
  };
  auto convertTiles = [&]() {
#pragma unroll
    for (int i = 0; i < 8; ++i) {
      hA[i][0] = (bf16_t)rawA[i].x; hA[i][1] = (bf16_t)rawA[i].y;
      hA[i][2] = (bf16_t)rawA[i].z; hA[i][3] = (bf16_t)rawA[i].w;
    }
#pragma unroll
    for (int j = 0; j < 8; ++j) {
      hB0[j] = (bf16_t)rawB[j].x;
      hB1[j] = (bf16_t)rawB[j].y;
    }
  };

  loadTiles(0);
  convertTiles();

  for (int it = 0; it < NITER; ++it) {
    __syncthreads();                       // previous compute done, LDS free

    // ---- store operand-ready bf16 tiles (regs already converted) ----
#pragma unroll
    for (int i = 0; i < 8; ++i)
      *(v4bf*)(sA + i * TSZ + aL * LSTR + ae0) = hA[i];  // ds_store_b64
    *(v8bf*)sBp0 = hB0;                                  // ds_store_b128
    *(v8bf*)sBp1 = hB1;                                  // ds_store_b128

    // ---- issue next-tile global loads; they complete during compute ----
    if (it + 1 < NITER) loadTiles(it + 1);

    __syncthreads();                       // staged tiles visible

    // ---- preload all operands, then 8 back-to-back WMMAs ----
    const bf16_t* bbase = sB + (2 * nh) * TSZ + lane * LSTR;
    v16bf bv0 = ldsOperand(bbase);
    v16bf bv1 = ldsOperand(bbase + TSZ);
    v16bf av[4];
#pragma unroll
    for (int a = 0; a < 4; ++a)
      av[a] = ldsOperand(sA + (4 * mh + a) * TSZ + lane * LSTR);
#pragma unroll
    for (int a = 0; a < 4; ++a) {
      acc[2 * a + 0] = __builtin_amdgcn_wmma_f32_16x16x32_bf16(
          false, av[a], false, bv0, (short)0, acc[2 * a + 0], false, false);
      acc[2 * a + 1] = __builtin_amdgcn_wmma_f32_16x16x32_bf16(
          false, av[a], false, bv1, (short)0, acc[2 * a + 1], false, false);
    }

    // ---- convert next tiles AFTER compute (loads had compute to land) ----
    if (it + 1 < NITER) convertTiles();
  }

  // ---- split-K reduction; C layout: VGPR j -> M = j + 8*hi, N = l15 ----
#pragma unroll
  for (int a = 0; a < 4; ++a) {
#pragma unroll
    for (int nb = 0; nb < 2; ++nb) {
#pragma unroll
      for (int j = 0; j < 8; ++j) {
        int m = m0 + (mh * 4 + a) * 16 + j + (hi ? 8 : 0);
        int n = (nh * 2 + nb) * 16 + l15;
        unsafeAtomicAdd(out + (size_t)m * DDIM + n, acc[2 * a + nb][j]);
      }
    }
  }
}

extern "C" void kernel_launch(void* const* d_in, const int* in_sizes, int n_in,
                              void* d_out, int out_size, void* d_ws, size_t ws_size,
                              hipStream_t stream) {
  const float* mask = (const float*)d_in[0];   // [1024, 131072] f32
  const float* x    = (const float*)d_in[1];   // [131072, 64] f32
  float* out = (float*)d_out;                  // [1024, 64] f32

  regionAgg_zero<<<dim3(128), dim3(128), 0, stream>>>(out);
  regionAgg_wmma<<<dim3((RREG / BM) * KSPLIT), dim3(128), 0, stream>>>(mask, x, out);
}
